// ROIViTExtractor_82712480186398
// MI455X (gfx1250) — matrix operand order repaired and checked by reading, not compiled, not run on hardware
//
#include <hip/hip_runtime.h>

// ---------------------------------------------------------------------------
// ROI ViT extractor for MI455X (gfx1250, wave32, WMMA, async-to-LDS).
//  Pipeline per block (32 patches x 768):
//   frames --global_load_async_to_lds_b128--> LDS f32 stage
//        --convert--> LDS bf16 A-tile --v_wmma_f32_16x16x32_bf16 vs bf16 W-->
//   token slab in LDS --ROI overlap reduce--> per-block partials (HBM)
//  finalize kernel sums 7 partials + writes cls row.
// ---------------------------------------------------------------------------

#define IMG 224
#define PS 16
#define HID 768
#define NS 14
#define NP 196
#define BATCH 64
#define NROI 6
#define MBLK 32            // patches per block
#define PTILES 7           // ceil(196/32)
#define NTILES 48          // 768/16
#define KBLK 24            // 768/32
#define AROW (HID + 8)     // bf16 LDS row pad (16B) to spread banks

// LDS partition (dynamic shared memory)
#define STAGE_BYTES (MBLK * HID * 4)            // 98304  (f32 stage / token slab)
#define AS_OFF      STAGE_BYTES
#define AS_BYTES    (MBLK * AROW * 2)           // 49664  (bf16 A tile)
#define MS_OFF      (AS_OFF + AS_BYTES)
#define MS_BYTES    (NROI * MBLK * 4)           // 768    (overlap coefficients)
#define SMEM_BYTES  (MS_OFF + MS_BYTES)         // 148736

typedef __attribute__((ext_vector_type(16))) __bf16 v16bf;
typedef __attribute__((ext_vector_type(8)))  float  v8f;

__device__ __forceinline__ unsigned short f2bf(float f) {
  union { float f; unsigned int u; } v; v.f = f;
  unsigned int r = v.u + 0x7FFFu + ((v.u >> 16) & 1u);   // round-nearest-even
  return (unsigned short)(r >> 16);
}

// ---------------------------------------------------------------------------
// Kernel 0: repack W (fp32 [k][n]) into WMMA B-fragment order, bf16.
// Wb[nt][kb][lane][16]: lane l -> n = nt*16 + (l&15), k = kb*32 + (l>>4)*16 + j.
// ---------------------------------------------------------------------------
__global__ void __launch_bounds__(256)
wconv_kernel(const float* __restrict__ W, unsigned short* __restrict__ Wb) {
  int idx = blockIdx.x * blockDim.x + threadIdx.x;     // 48*24*32*16 = 589824
  int j  = idx & 15;
  int l  = (idx >> 4) & 31;
  int kb = (idx >> 9) % KBLK;
  int nt = idx / (512 * KBLK);
  int k = kb * 32 + ((l >> 4) << 4) + j;
  int n = nt * 16 + (l & 15);
  Wb[idx] = f2bf(W[k * HID + n]);
}

// ---------------------------------------------------------------------------
// Kernel 1: async patch staging + bf16 GEMM + fused ROI partial reduction.
// grid = (PTILES, BATCH), block = 256 (8 waves). Wave owns 6 N-tiles x 2 M-tiles.
// ---------------------------------------------------------------------------
__global__ void __launch_bounds__(256)
patch_gemm_roi_kernel(const float* __restrict__ frames,
                      const unsigned short* __restrict__ Wb,
                      const float* __restrict__ bias,
                      const float* __restrict__ pos,
                      const float* __restrict__ bboxes,
                      float* __restrict__ partials) {
  extern __shared__ char smem[];
  float*          stage = reinterpret_cast<float*>(smem);            // 32x768 f32
  unsigned short* As    = reinterpret_cast<unsigned short*>(smem + AS_OFF);
  float*          Ms    = reinterpret_cast<float*>(smem + MS_OFF);   // 6x32

  const int blk = blockIdx.x;          // M-block of 32 patches
  const int b   = blockIdx.y;
  const int tid = threadIdx.x;

  // ---- Stage 1: async copy patch rows (16B chunks) into LDS f32 stage -----
  // i enumerates 32 patches x 192 chunks of 4 floats (d = ch*256 + y*16 + x).
  for (int i = tid; i < MBLK * 192; i += 256) {
    int mi = i / 192;
    int p  = blk * MBLK + mi;
    if (p < NP) {
      int d  = (i - mi * 192) * 4;
      int r  = p / NS, c = p - r * NS;
      int ch = d >> 8;
      int y  = (d >> 4) & 15;
      int x  = d & 15;
      const float* gp =
          frames + ((size_t)((b * 3 + ch) * IMG + (r * PS + y)) * IMG + c * PS + x);
      unsigned lds_addr = (unsigned)(uintptr_t)(stage + mi * HID + d);
      asm volatile("global_load_async_to_lds_b128 %0, %1, off"
                   :: "v"(lds_addr), "v"((unsigned long long)(uintptr_t)gp)
                   : "memory");
    }
  }
  asm volatile("s_wait_asynccnt 0x0" ::: "memory");
  __syncthreads();

  // ---- Stage 2: f32 -> bf16 A tile (zero-pad rows beyond patch 195) -------
  for (int i = tid; i < MBLK * HID; i += 256) {
    int mi = i / HID;
    int d  = i - mi * HID;
    int p  = blk * MBLK + mi;
    As[mi * AROW + d] = (p < NP) ? f2bf(stage[mi * HID + d]) : (unsigned short)0;
  }
  __syncthreads();

  // ---- Stage 3: GEMM. Each wave: 2 M-tiles x 6 N-tiles ---------------------
  const int wave = tid >> 5;
  const int lane = tid & 31;
  const int mrow = lane & 15;
  const int half = lane >> 4;

  v8f acc[2][6];
  #pragma unroll
  for (int a = 0; a < 2; ++a)
    #pragma unroll
    for (int t = 0; t < 6; ++t)
      acc[a][t] = (v8f){0.f, 0.f, 0.f, 0.f, 0.f, 0.f, 0.f, 0.f};

  for (int kb = 0; kb < KBLK; ++kb) {
    // Batch all 6 B fragments first -> one load clause, one wait, WMMA burst.
    v16bf bfr[6];
    #pragma unroll
    for (int t = 0; t < 6; ++t) {
      const int nt = wave + t * 8;
      bfr[t] = *reinterpret_cast<const v16bf*>(
          Wb + ((size_t)(nt * KBLK + kb) * 32 + lane) * 16);
    }
    // A fragments for both M-tiles (16x32 bf16 layout, K split by lane half).
    union { uint4 q[2]; v16bf v; } af[2];
    #pragma unroll
    for (int a = 0; a < 2; ++a) {
      const unsigned short* arow =
          &As[(a * 16 + mrow) * AROW + kb * 32 + half * 16];
      af[a].q[0] = *reinterpret_cast<const uint4*>(arow);
      af[a].q[1] = *reinterpret_cast<const uint4*>(arow + 8);
    }
    #pragma unroll
    for (int a = 0; a < 2; ++a)
      #pragma unroll
      for (int t = 0; t < 6; ++t)
        acc[a][t] = __builtin_amdgcn_wmma_f32_16x16x32_bf16(
            false, af[a].v, false, bfr[t], (short)0, acc[a][t], false, false);
  }

  // ---- Stage 4: epilogue -> token slab in LDS (reuse stage buffer) --------
  __syncthreads();   // everyone done reading stage (convert) before overwrite
  #pragma unroll
  for (int t = 0; t < 6; ++t) {
    const int n  = (wave + t * 8) * 16 + mrow;
    const float bn = bias[n];
    #pragma unroll
    for (int a = 0; a < 2; ++a) {
      #pragma unroll
      for (int v = 0; v < 8; ++v) {
        const int m = a * 16 + v + half * 8;
        const int p = blk * MBLK + m;
        stage[m * HID + n] =
            (p < NP) ? acc[a][t][v] + bn + pos[(size_t)(p + 1) * HID + n] : 0.0f;
      }
    }
  }

  // Overlap coefficients for this block's 32 patches (6 ROIs).
  if (tid < NROI * MBLK) {
    int n = tid / MBLK, m = tid - n * MBLK;
    int p = blk * MBLK + m;
    float coef = 0.0f;
    if (p < NP) {
      const float* bb = &bboxes[((size_t)b * NROI + n) * 4];
      int r = p / NS, c = p - r * NS;
      float px1 = (float)(c * PS), py1 = (float)(r * PS);
      float iw = fminf(bb[2], px1 + PS) - fmaxf(bb[0], px1);
      float ih = fminf(bb[3], py1 + PS) - fmaxf(bb[1], py1);
      coef = fmaxf(iw, 0.0f) * fmaxf(ih, 0.0f) * (1.0f / (float)(PS * PS));
    }
    Ms[n * MBLK + m] = coef;
  }
  __syncthreads();

  // ---- Stage 5: ROI partial reduction (token slab stays in LDS) -----------
  #pragma unroll
  for (int j = 0; j < 3; ++j) {
    const int h = tid + j * 256;
    float r0 = 0.f, r1 = 0.f, r2 = 0.f, r3 = 0.f, r4 = 0.f, r5 = 0.f;
    for (int m = 0; m < MBLK; ++m) {
      float tv = stage[m * HID + h];
      r0 = fmaf(Ms[0 * MBLK + m], tv, r0);
      r1 = fmaf(Ms[1 * MBLK + m], tv, r1);
      r2 = fmaf(Ms[2 * MBLK + m], tv, r2);
      r3 = fmaf(Ms[3 * MBLK + m], tv, r3);
      r4 = fmaf(Ms[4 * MBLK + m], tv, r4);
      r5 = fmaf(Ms[5 * MBLK + m], tv, r5);
    }
    float* pp = partials + (((size_t)b * PTILES + blk) * NROI) * HID + h;
    pp[0 * HID] = r0; pp[1 * HID] = r1; pp[2 * HID] = r2;
    pp[3 * HID] = r3; pp[4 * HID] = r4; pp[5 * HID] = r5;
  }
}

// ---------------------------------------------------------------------------
// Kernel 2: sum the 7 per-block partials, scale by 1/196; write cls row.
// ---------------------------------------------------------------------------
__global__ void __launch_bounds__(256)
finalize_kernel(const float* __restrict__ partials,
                const float* __restrict__ cls_tok,
                const float* __restrict__ pos,
                float* __restrict__ out) {
  const int b   = blockIdx.x;
  const int tid = threadIdx.x;
  const float inv = 1.0f / (float)NP;
  #pragma unroll
  for (int j = 0; j < 3; ++j) {
    const int h = tid + j * 256;
    out[(size_t)(b * 7) * HID + h] = cls_tok[h] + pos[h];   // cls + pos_emb[0]
    #pragma unroll
    for (int n = 0; n < NROI; ++n) {
      float s = 0.0f;
      for (int blk = 0; blk < PTILES; ++blk)
        s += partials[(((size_t)b * PTILES + blk) * NROI + n) * HID + h];
      out[(size_t)(b * 7 + 1 + n) * HID + h] = s * inv;
    }
  }
}

// ---------------------------------------------------------------------------
extern "C" void kernel_launch(void* const* d_in, const int* in_sizes, int n_in,
                              void* d_out, int out_size, void* d_ws, size_t ws_size,
                              hipStream_t stream) {
  (void)in_sizes; (void)n_in; (void)out_size; (void)ws_size;

  const float* frames  = (const float*)d_in[0];
  const float* bboxes  = (const float*)d_in[1];
  const float* W       = (const float*)d_in[2];
  const float* b_patch = (const float*)d_in[3];
  const float* cls     = (const float*)d_in[4];
  const float* pos     = (const float*)d_in[5];
  float* out = (float*)d_out;

  // Workspace: [bf16 repacked W (1.18 MB) | ROI partials (8.25 MB)]
  const size_t WB_ELEMS = (size_t)NTILES * KBLK * 32 * 16;   // 589824
  unsigned short* Wb = (unsigned short*)d_ws;
  float* partials = (float*)((char*)d_ws + WB_ELEMS * sizeof(unsigned short));

  wconv_kernel<<<dim3((unsigned)(WB_ELEMS / 256)), dim3(256), 0, stream>>>(W, Wb);
  patch_gemm_roi_kernel<<<dim3(PTILES, BATCH), dim3(256), SMEM_BYTES, stream>>>(
      frames, Wb, b_patch, pos, bboxes, partials);
  finalize_kernel<<<dim3(BATCH), dim3(256), 0, stream>>>(partials, cls, pos, out);
}
